// GRUCell_15401752723862
// MI455X (gfx1250) — compile-verified, matrix operand order checked by reading, not run on hardware
//
#include <hip/hip_runtime.h>
#include <hip/hip_bf16.h>

// ---------------------------------------------------------------------------
// GRU (B=32, T=1024, D=512, U=512), fp32 in/out.
// Phase 1: one big WMMA GEMM  x[32768,512] @ Wcat[512,1536] -> xpre (f32),
//          two N-tiles per wave (A-fragment reuse, 2 independent WMMA chains).
// Phase 2: persistent 16-workgroup kernel (1 per WGP, 288KB dynamic LDS):
//   - U_z/U_r/U_s fragment slices preloaded into LDS once (constant all steps)
//   - per step: h staged to LDS via global_load_async_to_lds_b128 (ASYNCcnt)
//     A) z,r tiles: h @ [U_z|U_r] (128 wave-tiles), sigmoid, build r*h (bf16)
//     -- grid sync --
//     B) s tiles: (r*h) @ U_s (64 wave-tiles), tanh, h update + output
//     -- grid sync --
// WMMA: v_wmma_f32_16x16x32_bf16 with split accumulators (2x ILP).
// ---------------------------------------------------------------------------

typedef __attribute__((ext_vector_type(16))) __bf16 v16bf;
typedef __attribute__((ext_vector_type(8)))  __bf16 v8bf;
typedef __attribute__((ext_vector_type(8)))  float  v8f;

#define BATCH 32
#define SEQ   1024
#define NCOL  1536           // 3 gates * 512
#define KT    16             // 512 / 32
#define NWG_REC 16           // persistent workgroups (1 per WGP)

// dynamic LDS layout (bytes)
#define LDS_UA    0          // 128 KB: phase-A U slices  (8 waves x 16 KB)
#define LDS_UB    131072     // 128 KB: phase-B U slices  (blocks 0..7)
#define LDS_STAGE 262144     // 32 KB : h / r*h bf16 staging [32][512]
#define LDS_TOTAL 294912     // 288 KB <= 320 KB per WGP

#define WMMA_BF16(A, B, C) \
  __builtin_amdgcn_wmma_f32_16x16x32_bf16(false, (A), false, (B), (short)0, (C), false, false)

// ---------------- fragment helpers ----------------

// A-fragment (16x32 bf16) from row-major [32][512] bf16 matrix (LDS).
// lane<16 -> M=lane, K=kt*32+{0..7,16..23}; lane>=16 -> M=lane-16, K=+8.
__device__ __forceinline__ v16bf load_afrag(const __bf16* m, int mt, int kt, int lane) {
  int M  = mt * 16 + (lane & 15);
  int kb = kt * 32 + ((lane >> 4) & 1) * 8;
  const __bf16* row = m + M * 512 + kb;
  v8bf lo = *(const v8bf*)(row);
  v8bf hi = *(const v8bf*)(row + 16);
  return __builtin_shufflevector(lo, hi, 0,1,2,3,4,5,6,7,8,9,10,11,12,13,14,15);
}

// B-fragment from packed global array [kt][ntile_global(96)][lane][16]
__device__ __forceinline__ v16bf load_bfrag_g(const __bf16* frag, int kt, int ntg, int lane) {
  return ((const v16bf*)frag)[((size_t)(kt * 96 + ntg)) * 32 + lane];
}

// B-fragment from block-local LDS cache [slot(8)][kt][lane][16]
__device__ __forceinline__ v16bf load_bfrag_lds(const __bf16* base, int slot, int kt, int lane) {
  return ((const v16bf*)base)[(slot * 16 + kt) * 32 + lane];
}

__device__ __forceinline__ float sigmoidf_(float x) {
  return 1.0f / (1.0f + __expf(-x));
}

// 32KB global -> LDS via gfx1250 async-to-LDS path (ASYNCcnt tracked).
// Each lane issues 8x 16B transfers; wait own ASYNCcnt, then caller barriers.
__device__ __forceinline__ void async_stage32x512(uint32_t lds_off, const void* src, int tid) {
  #pragma unroll
  for (int i = 0; i < 8; ++i) {
    uint32_t off = (uint32_t)((tid + 256 * i) * 16);
    asm volatile("global_load_async_to_lds_b128 %0, %1, %2"
                 :: "v"(lds_off + off), "v"(off), "s"(src)
                 : "memory");
  }
  asm volatile("s_wait_asynccnt 0x0" ::: "memory");
}

__device__ __forceinline__ void grid_sync(unsigned* counter, unsigned* gen) {
  __syncthreads();
  if (threadIdx.x == 0) {
    __threadfence();
    unsigned g    = __hip_atomic_load(gen, __ATOMIC_RELAXED, __HIP_MEMORY_SCOPE_AGENT);
    unsigned prev = __hip_atomic_fetch_add(counter, 1u, __ATOMIC_ACQ_REL, __HIP_MEMORY_SCOPE_AGENT);
    if (prev == NWG_REC - 1u) {
      __hip_atomic_store(counter, 0u, __ATOMIC_RELAXED, __HIP_MEMORY_SCOPE_AGENT);
      __hip_atomic_fetch_add(gen, 1u, __ATOMIC_RELEASE, __HIP_MEMORY_SCOPE_AGENT);
    } else {
      while (__hip_atomic_load(gen, __ATOMIC_ACQUIRE, __HIP_MEMORY_SCOPE_AGENT) == g) {
        __builtin_amdgcn_s_sleep(8);
      }
    }
  }
  __syncthreads();
}

// ---------------- prep kernels ----------------

__global__ void pack_b_frags(const float* __restrict__ Wz,
                             const float* __restrict__ Wr,
                             const float* __restrict__ Ws,
                             __bf16* __restrict__ frag) {
  int idx = blockIdx.x * 256 + threadIdx.x;          // 16*96*32 = 49152
  if (idx >= 16 * 96 * 32) return;
  int lane = idx & 31;
  int ntg  = (idx >> 5) % 96;
  int kt   = idx / (96 * 32);
  int g    = ntg / 32;
  int ntl  = ntg % 32;
  const float* W = (g == 0) ? Wz : (g == 1) ? Wr : Ws;
  int N  = ntl * 16 + (lane & 15);
  int hb = ((lane >> 4) & 1) * 8;
  __bf16* o = frag + ((size_t)idx) * 16;             // idx == ((kt*96+ntg)*32+lane)
  #pragma unroll
  for (int j = 0; j < 8; ++j) o[j]     = (__bf16)W[(kt * 32 + hb + j) * 512 + N];
  #pragma unroll
  for (int j = 0; j < 8; ++j) o[8 + j] = (__bf16)W[(kt * 32 + 16 + hb + j) * 512 + N];
}

__global__ void pack_x_frags(const float* __restrict__ x, __bf16* __restrict__ frag) {
  int idx  = blockIdx.x * 256 + threadIdx.x;         // 2048*16*32 = 1048576
  int lane = idx & 31;
  int kt   = (idx >> 5) & 15;
  int rt   = idx >> 9;
  int row  = rt * 16 + (lane & 15);
  int kb   = kt * 32 + ((lane >> 4) & 1) * 8;
  const float* r = x + (size_t)row * 512;
  __bf16* o = frag + (size_t)idx * 16;
  #pragma unroll
  for (int j = 0; j < 8; ++j) o[j]     = (__bf16)r[kb + j];
  #pragma unroll
  for (int j = 0; j < 8; ++j) o[8 + j] = (__bf16)r[kb + 16 + j];
}

__global__ void init_state(float* __restrict__ hf32, __bf16* __restrict__ hbf,
                           unsigned* __restrict__ sync) {
  int i = blockIdx.x * 256 + threadIdx.x;            // 16384 threads
  if (i < BATCH * 512) { hf32[i] = 0.0f; hbf[i] = (__bf16)0.0f; }
  if (i < 64) sync[i] = 0u;
}

// ---------------- phase 1: input projections ----------------
// Each wave: one row-tile, TWO column tiles (A reuse, 2 independent chains).
__global__ void __launch_bounds__(256) input_gemm(
    const __bf16* __restrict__ xfrag, const __bf16* __restrict__ wfrag,
    const float* __restrict__ bz, const float* __restrict__ br,
    const float* __restrict__ bs, float* __restrict__ xpre) {
  int lane = threadIdx.x & 31;
  int wv   = threadIdx.x >> 5;
  int task = blockIdx.x * 8 + wv;                    // 2048*48 = 98304 tasks
  int rt   = task / 48;
  int npr  = task % 48;
  int ntg0 = npr * 2;
  int ntg1 = npr * 2 + 1;

  v8f acc0 = {}, acc1 = {};
  #pragma unroll
  for (int kt = 0; kt < KT; ++kt) {
    v16bf a  = ((const v16bf*)xfrag)[((size_t)(rt * 16 + kt)) * 32 + lane];
    v16bf b0 = load_bfrag_g(wfrag, kt, ntg0, lane);
    v16bf b1 = load_bfrag_g(wfrag, kt, ntg1, lane);
    acc0 = WMMA_BF16(a, b0, acc0);
    acc1 = WMMA_BF16(a, b1, acc1);
  }

  int hi = (lane >> 4) & 1;
  #pragma unroll
  for (int tile = 0; tile < 2; ++tile) {
    int col = (tile == 0 ? ntg0 : ntg1) * 16 + (lane & 15);
    int g   = col >> 9;
    int u   = col & 511;
    float bias = ((g == 0) ? bz : (g == 1) ? br : bs)[u];
    v8f acc = (tile == 0) ? acc0 : acc1;
    #pragma unroll
    for (int i = 0; i < 8; ++i) {
      int row = rt * 16 + i + 8 * hi;                // row = b*T + t
      int b   = row >> 10;
      int t   = row & 1023;
      xpre[((size_t)(t * BATCH + b)) * NCOL + col] = acc[i] + bias;
    }
  }
}

// ---------------- phase 2: persistent recurrence ----------------
__global__ void __launch_bounds__(256, 1) gru_recurrence(
    const __bf16* __restrict__ ufrag, const float* __restrict__ xpre,
    float* __restrict__ hf32, __bf16* __restrict__ hbf,
    float* __restrict__ zbuf, __bf16* __restrict__ rhbuf,
    float* __restrict__ out, unsigned* __restrict__ sync) {
  extern __shared__ char dynsm[];
  __bf16* uA    = (__bf16*)(dynsm + LDS_UA);
  __bf16* uB    = (__bf16*)(dynsm + LDS_UB);
  __bf16* stage = (__bf16*)(dynsm + LDS_STAGE);
  uint32_t stage_off = (uint32_t)(uintptr_t)stage;   // LDS byte offset (addr[31:0])

  int tid  = threadIdx.x;
  int lane = tid & 31;
  int wv   = tid >> 5;
  int blk  = blockIdx.x;
  int gtile = blk * 8 + wv;                          // 0..127

  // ---- one-time preload of this block's constant U fragment slices ----
  int gA = (blk >= 8) ? 1 : 0;                       // blocks 0-7: z; 8-15: r
  for (int idx = tid; idx < 8 * 16 * 64; idx += 256) {
    int chunk = idx & 63, kt = (idx >> 6) & 15, s = idx >> 10;
    int ntg = gA * 32 + ((blk * 8 + s) & 31);
    ((v8bf*)uA)[(s * 16 + kt) * 64 + chunk] =
        ((const v8bf*)ufrag)[(size_t)(kt * 96 + ntg) * 64 + chunk];
  }
  if (blk < 8) {                                     // blocks 0-7 also run phase B (s gate)
    for (int idx = tid; idx < 8 * 16 * 64; idx += 256) {
      int chunk = idx & 63, kt = (idx >> 6) & 15, s = idx >> 10;
      int ntg = 64 + ((blk * 8 + s) & 31);
      ((v8bf*)uB)[(s * 16 + kt) * 64 + chunk] =
          ((const v8bf*)ufrag)[(size_t)(kt * 96 + ntg) * 64 + chunk];
    }
  }
  __syncthreads();

  for (int t = 0; t < SEQ; ++t) {
    // ---- Phase A: z and r tiles (h @ [U_z | U_r]) ----
    async_stage32x512(stage_off, hbf, tid);          // bf16 h -> LDS (ASYNCcnt)
    __syncthreads();
    {
      int g   = gtile >> 6;                          // 0 = z, 1 = r
      int rem = gtile & 63;
      int mt  = rem >> 5;
      int nt  = rem & 31;
      v8f acc0 = {}, acc1 = {};
      #pragma unroll
      for (int kt = 0; kt < KT; kt += 2) {           // two independent chains
        v16bf a0 = load_afrag(stage, mt, kt, lane);
        v16bf b0 = load_bfrag_lds(uA, wv, kt, lane);
        acc0 = WMMA_BF16(a0, b0, acc0);
        v16bf a1 = load_afrag(stage, mt, kt + 1, lane);
        v16bf b1 = load_bfrag_lds(uA, wv, kt + 1, lane);
        acc1 = WMMA_BF16(a1, b1, acc1);
      }
      v8f acc = acc0 + acc1;
      int hi = (lane >> 4) & 1;
      int u  = nt * 16 + (lane & 15);
      #pragma unroll
      for (int i = 0; i < 8; ++i) {
        int b = mt * 16 + i + 8 * hi;                // batch row
        float pre = xpre[((size_t)(t * BATCH + b)) * NCOL + g * 512 + u] + acc[i];
        float v   = sigmoidf_(pre);
        if (g == 0) {
          zbuf[b * 512 + u] = v;                     // z gate
        } else {
          float hv = hf32[b * 512 + u];              // r gate -> r*h (bf16)
          rhbuf[b * 512 + u] = (__bf16)(v * hv);
        }
      }
    }
    grid_sync(&sync[0], &sync[1]);

    // ---- Phase B: s tiles ((r*h) @ U_s), gate update, output ----
    async_stage32x512(stage_off, rhbuf, tid);
    __syncthreads();
    if (gtile < 64) {
      int mt = gtile >> 5;
      int nt = gtile & 31;
      v8f acc0 = {}, acc1 = {};
      #pragma unroll
      for (int kt = 0; kt < KT; kt += 2) {
        v16bf a0 = load_afrag(stage, mt, kt, lane);
        v16bf b0 = load_bfrag_lds(uB, wv, kt, lane);
        acc0 = WMMA_BF16(a0, b0, acc0);
        v16bf a1 = load_afrag(stage, mt, kt + 1, lane);
        v16bf b1 = load_bfrag_lds(uB, wv, kt + 1, lane);
        acc1 = WMMA_BF16(a1, b1, acc1);
      }
      v8f acc = acc0 + acc1;
      int hi = (lane >> 4) & 1;
      int u  = nt * 16 + (lane & 15);
      #pragma unroll
      for (int i = 0; i < 8; ++i) {
        int b = mt * 16 + i + 8 * hi;
        float pre = xpre[((size_t)(t * BATCH + b)) * NCOL + 1024 + u] + acc[i];
        float s   = tanhf(pre);
        float z   = zbuf[b * 512 + u];
        float hv  = hf32[b * 512 + u];
        float hn  = (1.0f - z) * hv + z * s;
        hf32[b * 512 + u] = hn;
        hbf[b * 512 + u]  = (__bf16)hn;
        out[((size_t)b * SEQ + t) * 512 + u] = hn;   // [B,T,U]
      }
    }
    grid_sync(&sync[2], &sync[3]);
  }
}

// ---------------- launcher ----------------
extern "C" void kernel_launch(void* const* d_in, const int* in_sizes, int n_in,
                              void* d_out, int out_size, void* d_ws, size_t ws_size,
                              hipStream_t stream) {
  const float* x  = (const float*)d_in[0];
  const float* Wz = (const float*)d_in[1];
  const float* Uz = (const float*)d_in[2];
  const float* bz = (const float*)d_in[3];
  const float* Wr = (const float*)d_in[4];
  const float* Ur = (const float*)d_in[5];
  const float* br = (const float*)d_in[6];
  const float* Ws = (const float*)d_in[7];
  const float* Us = (const float*)d_in[8];
  const float* bs = (const float*)d_in[9];
  float* out = (float*)d_out;

  char* p = (char*)d_ws;
  __bf16* wfrag = (__bf16*)p;  p += (size_t)16 * 96 * 32 * 16 * 2;     // 1.5 MB
  __bf16* ufrag = (__bf16*)p;  p += (size_t)16 * 96 * 32 * 16 * 2;     // 1.5 MB
  __bf16* xfrag = (__bf16*)p;  p += (size_t)2048 * 16 * 32 * 16 * 2;   // 32 MB
  float*  xpre  = (float*)p;   p += (size_t)32768 * 1536 * 4;          // 192 MB
  float*  hf32  = (float*)p;   p += (size_t)BATCH * 512 * 4;
  __bf16* hbf   = (__bf16*)p;  p += (size_t)BATCH * 512 * 2;
  float*  zbuf  = (float*)p;   p += (size_t)BATCH * 512 * 4;
  __bf16* rhbuf = (__bf16*)p;  p += (size_t)BATCH * 512 * 2;
  unsigned* sync = (unsigned*)p;

  pack_b_frags<<<192, 256, 0, stream>>>(Wz, Wr, Ws, wfrag);
  pack_b_frags<<<192, 256, 0, stream>>>(Uz, Ur, Us, ufrag);
  pack_x_frags<<<4096, 256, 0, stream>>>(x, xfrag);
  init_state<<<64, 256, 0, stream>>>(hf32, hbf, sync);

  // phase 1: xpre = x @ [Wz|Wr|Ws] + b   (98304 wave-tasks, 8 per block)
  input_gemm<<<12288, 256, 0, stream>>>(xfrag, wfrag, bz, br, bs, xpre);

  // phase 2: persistent recurrence, 288KB dynamic LDS per workgroup
  gru_recurrence<<<NWG_REC, 256, LDS_TOTAL, stream>>>(ufrag, xpre, hf32, hbf,
                                                      zbuf, rhbuf, out, sync);
}
